// MultiReadout_35107062678103
// MI455X (gfx1250) — matrix operand order repaired and checked by reading, not compiled, not run on hardware
//
#include <hip/hip_runtime.h>
#include <hip/hip_bf16.h>

#define NN 1000000
#define HH 128
#define BB 4096

typedef __attribute__((ext_vector_type(16))) _Float16 v16h;
typedef __attribute__((ext_vector_type(8)))  float    v8f;

#define ROWS 96          // rows of x staged per block (6 M-tiles of 16)
#define PITCH 132        // floats per LDS row (528B = 33*16B, bank-friendly)

// ---- CDNA5 async global->LDS copy (ASYNCcnt path) -------------------------
__device__ __forceinline__ void async_copy_b128(unsigned lds_off, const void* gptr) {
    asm volatile("global_load_async_to_lds_b128 %0, %1, off"
                 :: "v"(lds_off), "v"(gptr) : "memory");
}
__device__ __forceinline__ void wait_async0() {
    asm volatile("s_wait_asynccnt 0" ::: "memory");
}

// ---- order-preserving float<->uint encoding for atomic max ----------------
__device__ __forceinline__ unsigned fenc(float f) {
    unsigned u = __float_as_uint(f);
    return (u & 0x80000000u) ? ~u : (u | 0x80000000u);
}
__device__ __forceinline__ float fdec(unsigned k) {
    unsigned u = (k & 0x80000000u) ? (k & 0x7FFFFFFFu) : ~k;
    return __uint_as_float(u);
}

// ---------------------------------------------------------------------------
__global__ void k_init(float* __restrict__ sumpool, unsigned* __restrict__ maxenc,
                       float* __restrict__ attbuf, float* __restrict__ counts,
                       float* __restrict__ smax, float* __restrict__ sumexp) {
    int i = blockIdx.x * blockDim.x + threadIdx.x;
    if (i < BB * HH) {
        sumpool[i] = 0.f;
        maxenc[i]  = 0x007FFFFFu;   // enc(-inf)
        attbuf[i]  = 0.f;
    }
    if (i < BB) { counts[i] = 0.f; smax[i] = 0.f; sumexp[i] = 0.f; }
}

// ---------------------------------------------------------------------------
// K1: one pass over x. GEMM scores via WMMA f16 + tanh + w2 dot, plus
// segment sum/max/count pooling from the LDS-staged tile.
__global__ __launch_bounds__(256) void k_scores_pool(
    const float* __restrict__ x, const int* __restrict__ batch,
    const float* __restrict__ W1, const float* __restrict__ b1,
    const float* __restrict__ w2,
    float* __restrict__ scores, float* __restrict__ smax,
    float* __restrict__ sumpool, unsigned* __restrict__ maxenc,
    float* __restrict__ counts)
{
    __shared__ float lds_x[ROWS * PITCH];   // ~50.7 KB
    __shared__ float lds_score[ROWS];
    __shared__ int   lds_seg[ROWS];

    const int t = threadIdx.x;
    const long long base = (long long)blockIdx.x * ROWS;

    // -- phase 0: async-stage x tile into LDS (16B chunks, 32 per row) ------
    #pragma unroll
    for (int it = 0; it < (ROWS * 32) / 256; ++it) {   // 12 iterations
        int chunk = t + it * 256;
        int row = chunk >> 5;
        int c   = chunk & 31;
        long long grow = base + row; if (grow >= NN) grow = NN - 1;
        const float* gp = x + grow * (long long)HH + c * 4;
        unsigned loff = (unsigned)(size_t)&lds_x[row * PITCH + c * 4];
        async_copy_b128(loff, gp);
    }
    if (t < ROWS) {
        lds_score[t] = 0.f;
        long long grow = base + t; if (grow >= NN) grow = NN - 1;
        lds_seg[t] = batch[grow];
    }
    wait_async0();
    __syncthreads();

    // -- phase 1: per-wave W1 fragments (wave w owns N columns 16w..16w+15) -
    const int wave  = t >> 5;
    const int lane  = t & 31;
    const int ln16  = lane & 15;
    const int lhalf = lane >> 4;
    const int ncol  = wave * 16 + ln16;

    v16h bfrag[4];
    #pragma unroll
    for (int ks = 0; ks < 4; ++ks) {
        const int kb = ks * 32 + lhalf * 16;   // B layout: contiguous K per half
        v16h bf;
        #pragma unroll
        for (int i = 0; i < 16; ++i)
            bf[i] = (_Float16)W1[(kb + i) * HH + ncol];
        bfrag[ks] = bf;
    }
    const float biasn = b1[ncol];
    const float w2n   = w2[ncol];

    // -- phase 2: 6 M-tiles of WMMA + tanh/w2 score reduction ---------------
    for (int mt = 0; mt < ROWS / 16; ++mt) {
        v8f c = {};
        #pragma unroll
        for (int ks = 0; ks < 4; ++ks) {
            const int kb = ks * 32 + lhalf * 16; // A layout: contiguous K per half {kb..kb+15}
            const float* ap = &lds_x[(mt * 16 + ln16) * PITCH + kb];
            v16h a;
            #pragma unroll
            for (int i = 0; i < 8; ++i) a[i]     = (_Float16)ap[i];
            #pragma unroll
            for (int i = 0; i < 8; ++i) a[8 + i] = (_Float16)ap[8 + i];
            c = __builtin_amdgcn_wmma_f32_16x16x32_f16(
                    false, a, false, bfrag[ks], (short)0, c, false, false);
        }
        #pragma unroll
        for (int v = 0; v < 8; ++v) {
            float tv = tanhf(c[v] + biasn) * w2n;
            tv += __shfl_xor(tv, 1, 16);
            tv += __shfl_xor(tv, 2, 16);
            tv += __shfl_xor(tv, 4, 16);
            tv += __shfl_xor(tv, 8, 16);
            if (ln16 == 0)
                atomicAdd(&lds_score[mt * 16 + v + 8 * lhalf], tv);
        }
    }
    __syncthreads();

    // -- phase 3: emit scores + clipped segment score-max -------------------
    if (t < ROWS) {
        long long grow = base + t;
        if (grow < NN) {
            float s = lds_score[t];
            scores[grow] = s;
            if (s > 0.f)
                atomicMax((int*)&smax[lds_seg[t]], __float_as_int(s));
        }
    }

    // -- phase 4: segment sum/max/count with run compression ----------------
    {
        const int f    = t & 127;
        const int half = t >> 7;
        float runsum = 0.f, runmax = 0.f, runcnt = 0.f;
        int curseg = -1;
        for (int r = half; r < ROWS; r += 2) {
            long long grow = base + r;
            if (grow >= NN) break;
            int seg   = lds_seg[r];
            float val = lds_x[r * PITCH + f];
            if (seg != curseg) {
                if (curseg >= 0) {
                    atomicAdd(&sumpool[curseg * HH + f], runsum);
                    atomicMax(&maxenc[curseg * HH + f], fenc(runmax));
                    if (f == 0) atomicAdd(&counts[curseg], runcnt);
                }
                curseg = seg; runsum = val; runmax = val; runcnt = 1.f;
            } else {
                runsum += val; runmax = fmaxf(runmax, val); runcnt += 1.f;
            }
        }
        if (curseg >= 0) {
            atomicAdd(&sumpool[curseg * HH + f], runsum);
            atomicMax(&maxenc[curseg * HH + f], fenc(runmax));
            if (f == 0) atomicAdd(&counts[curseg], runcnt);
        }
    }
}

// ---------------------------------------------------------------------------
// K2: exp(score - segmax) in place + segment sum of exp (wave-combined).
__global__ __launch_bounds__(256) void k_exp(
    const int* __restrict__ batch, float* __restrict__ scores,
    const float* __restrict__ smax, float* __restrict__ sumexp)
{
    long long i = (long long)blockIdx.x * 256 + threadIdx.x;
    bool valid = i < NN;
    long long ci = valid ? i : (NN - 1);
    int seg = batch[ci];
    float e = valid ? __expf(scores[ci] - smax[seg]) : 0.f;
    if (valid) scores[i] = e;

    int seg0 = __shfl(seg, 0, 32);
    if (__all(seg == seg0)) {
        float tot = e;
        tot += __shfl_xor(tot, 1, 32);
        tot += __shfl_xor(tot, 2, 32);
        tot += __shfl_xor(tot, 4, 32);
        tot += __shfl_xor(tot, 8, 32);
        tot += __shfl_xor(tot, 16, 32);
        if ((threadIdx.x & 31) == 0) atomicAdd(&sumexp[seg0], tot);
    } else if (valid) {
        atomicAdd(&sumexp[seg], e);
    }
}

// ---------------------------------------------------------------------------
// K3: attentive pool — second pass over x, run-compressed segment adds.
__global__ __launch_bounds__(256) void k_att(
    const float* __restrict__ x, const int* __restrict__ batch,
    const float* __restrict__ expbuf, const float* __restrict__ sumexp,
    float* __restrict__ attbuf)
{
    __shared__ float lds_norm[128];
    __shared__ int   lds_seg2[128];
    const int t = threadIdx.x;
    const long long base = (long long)blockIdx.x * 128;

    if (t < 128) {
        long long g = base + t;
        if (g < NN) {
            int seg = batch[g];
            lds_seg2[t] = seg;
            lds_norm[t] = expbuf[g] / sumexp[seg];
        } else { lds_seg2[t] = -1; lds_norm[t] = 0.f; }
    }
    __syncthreads();

    const int f = t & 127, half = t >> 7;
    float run = 0.f; int cur = -1;
    for (int r = half; r < 128; r += 2) {
        long long g = base + r;
        if (g >= NN) break;
        int seg = lds_seg2[r];
        float v = x[g * (long long)HH + f] * lds_norm[r];
        if (seg != cur) {
            if (cur >= 0) atomicAdd(&attbuf[cur * HH + f], run);
            cur = seg; run = v;
        } else run += v;
    }
    if (cur >= 0) atomicAdd(&attbuf[cur * HH + f], run);
}

// ---------------------------------------------------------------------------
// K4: weighted fusion (softmax of 3 pool weights) + LayerNorm over H=128.
__global__ __launch_bounds__(128) void k_fuse(
    const float* __restrict__ sumpool, const unsigned* __restrict__ maxenc,
    const float* __restrict__ attbuf, const float* __restrict__ counts,
    const float* __restrict__ pw, const float* __restrict__ gamma,
    const float* __restrict__ beta, float* __restrict__ out)
{
    __shared__ float red[8];
    const int s = blockIdx.x, f = threadIdx.x;

    float p0 = pw[0], p1 = pw[1], p2 = pw[2];
    float pm = fmaxf(p0, fmaxf(p1, p2));
    float e0 = __expf(p0 - pm), e1 = __expf(p1 - pm), e2 = __expf(p2 - pm);
    float inv = 1.f / (e0 + e1 + e2);

    float cnt  = counts[s];
    float mean = sumpool[s * HH + f] / cnt;
    float mx   = fdec(maxenc[s * HH + f]);
    float att  = attbuf[s * HH + f];
    float emb  = (e0 * mean + e1 * mx + e2 * att) * inv;

    float sum = emb, sq = emb * emb;
    for (int m = 16; m >= 1; m >>= 1) {
        sum += __shfl_xor(sum, m, 32);
        sq  += __shfl_xor(sq,  m, 32);
    }
    if ((f & 31) == 0) { red[f >> 5] = sum; red[4 + (f >> 5)] = sq; }
    __syncthreads();
    float ts = red[0] + red[1] + red[2] + red[3];
    float tq = red[4] + red[5] + red[6] + red[7];
    float mu  = ts * (1.f / HH);
    float var = tq * (1.f / HH) - mu * mu;

    out[s * HH + f] = (emb - mu) * rsqrtf(var + 1e-5f) * gamma[f] + beta[f];
}

// ---------------------------------------------------------------------------
extern "C" void kernel_launch(void* const* d_in, const int* in_sizes, int n_in,
                              void* d_out, int out_size, void* d_ws, size_t ws_size,
                              hipStream_t stream) {
    const float* x     = (const float*)d_in[0];
    const int*   batch = (const int*)  d_in[1];
    const float* W1    = (const float*)d_in[2];
    const float* b1    = (const float*)d_in[3];
    const float* w2    = (const float*)d_in[4];
    const float* pw    = (const float*)d_in[5];
    const float* gamma = (const float*)d_in[6];
    const float* beta  = (const float*)d_in[7];
    float* out = (float*)d_out;

    float*    ws      = (float*)d_ws;
    float*    sumpool = ws;                              // B*H
    unsigned* maxenc  = (unsigned*)(sumpool + BB * HH);  // B*H
    float*    attbuf  = (float*)(maxenc + BB * HH);      // B*H
    float*    counts  = attbuf + BB * HH;                // B
    float*    smaxv   = counts + BB;                     // B
    float*    sumexp  = smaxv + BB;                      // B
    float*    scores  = sumexp + BB;                     // N (reused as exp buf)

    k_init<<<(BB * HH + 255) / 256, 256, 0, stream>>>(
        sumpool, maxenc, attbuf, counts, smaxv, sumexp);

    int nblk1 = (NN + ROWS - 1) / ROWS;
    k_scores_pool<<<nblk1, 256, 0, stream>>>(
        x, batch, W1, b1, w2, scores, smaxv, sumpool, maxenc, counts);

    k_exp<<<(NN + 255) / 256, 256, 0, stream>>>(batch, scores, smaxv, sumexp);

    int nblk3 = (NN + 127) / 128;
    k_att<<<nblk3, 256, 0, stream>>>(x, batch, scores, sumexp, attbuf);

    k_fuse<<<BB, 128, 0, stream>>>(
        sumpool, maxenc, attbuf, counts, pw, gamma, beta, out);
}
